// E58LearnedRadiiCell_45792941310467
// MI455X (gfx1250) — matrix operand order repaired and checked by reading, not compiled, not run on hardware
//
#include <hip/hip_runtime.h>
#include <hip/hip_bf16.h>
#include <math.h>

// ---------------------------------------------------------------------------
// Problem constants (match reference setup_inputs)
// ---------------------------------------------------------------------------
#define TT 1024      // timesteps
#define BB 16        // batch
#define DD 1024      // hidden dim
#define BD (BB * DD) // 16384 elements per timestep
#define KC 32        // K-chunks of 32 (DD / 32)
#define NT 64        // N-tiles of 16 (DD / 16)
#define MAX_RADIUS 0.999f
#define EPS_F 1e-8f

#define SCAN_BLOCKS 16      // persistent blocks in the scan
#define SCAN_THREADS 128    // 4 waves -> 4 n-tiles per block, 64 total
#define LDS_PLANE (KC * 32 * 16)                // shorts per (wave, plane) = 16384
#define SCAN_LDS_BYTES (4 * 2 * LDS_PLANE * 2)  // 4 waves * hi/lo * 32KB = 256KB

typedef __attribute__((ext_vector_type(16))) __bf16 v16bf;
typedef __attribute__((ext_vector_type(8)))  float  v8f;
typedef __attribute__((ext_vector_type(4)))  unsigned int u32x4;
typedef __attribute__((ext_vector_type(8)))  int          i32x8;
typedef __attribute__((ext_vector_type(4)))  int          i32x4;

// TDM availability (this toolchain declares the 6-arg clang-23 form)
#if defined(__has_builtin)
#if __has_builtin(__builtin_amdgcn_tensor_load_to_lds) && __has_builtin(__builtin_amdgcn_s_wait_tensorcnt)
#define HAVE_TDM 1
#endif
#endif

// ---------------------------------------------------------------------------
// Workspace layout (bytes)
// ---------------------------------------------------------------------------
static constexpr size_t WS_BAR    = 0;
static constexpr size_t WS_SIGMA  = 64;
static constexpr size_t WS_WH_HI  = 1024;                                  // 2 MB
static constexpr size_t WS_WH_LO  = WS_WH_HI + (size_t)DD * DD * 2;        // 2 MB
static constexpr size_t WS_WX_HI  = WS_WH_LO + (size_t)DD * DD * 2;        // 2 MB
static constexpr size_t WS_WX_LO  = WS_WX_HI + (size_t)DD * DD * 2;        // 2 MB
static constexpr size_t WS_P      = WS_WX_LO + (size_t)DD * DD * 2;        // 64 MB
static constexpr size_t WS_XA_HI  = WS_P + (size_t)TT * BD * 4;            // 32 MB
static constexpr size_t WS_XA_LO  = WS_XA_HI + (size_t)TT * BD * 2;        // 32 MB
static constexpr size_t WS_HA     = WS_XA_LO + (size_t)TT * BD * 2;        // 4 * 32 KB

// ---------------------------------------------------------------------------
// bf16 helpers (RNE; hi/lo split -> ~16 mantissa bits, near-f32 products)
// ---------------------------------------------------------------------------
static __device__ __forceinline__ unsigned short f2bf(float f) {
    unsigned u = __float_as_uint(f);
    u += 0x7FFFu + ((u >> 16) & 1u);
    return (unsigned short)(u >> 16);
}
static __device__ __forceinline__ float bf2f(unsigned short h) {
    return __uint_as_float(((unsigned)h) << 16);
}

// Inverse A-layout map: element (row b, k') of a 16x32 bf16 A tile lives at
// lane = half*16 + b, elem e, with half = (k'>>3)&1, e = (k'&7) + ((k'>>4)<<3)
static __device__ __forceinline__ int a_pack_idx(int b, int n) {
    int kc   = n >> 5;
    int kp   = n & 31;
    int half = (kp >> 3) & 1;
    int e    = (kp & 7) + ((kp >> 4) << 3);
    return ((kc * 32 + half * 16 + b) << 4) + e;
}

// ---------------------------------------------------------------------------
// Kernel 1: init — zero barrier, h_all[0] = h0, pack h0 into A-buffer parity 0
// ---------------------------------------------------------------------------
__global__ __launch_bounds__(256) void k_init(const float* __restrict__ h0,
                                              float* __restrict__ hall,
                                              unsigned short* __restrict__ Ahi0,
                                              unsigned short* __restrict__ Alo0,
                                              unsigned* __restrict__ bar) {
    int idx = blockIdx.x * blockDim.x + threadIdx.x;
    if (idx == 0) *bar = 0u;
    if (idx < BD) {
        float v = h0[idx];
        hall[idx] = v;
        int b = idx >> 10;          // batch
        int n = idx & (DD - 1);     // hidden index -> next-step K index
        int ai = a_pack_idx(b, n);
        unsigned short h = f2bf(v);
        Ahi0[ai] = h;
        Alo0[ai] = f2bf(v - bf2f(h));
    }
}

// ---------------------------------------------------------------------------
// Kernel 2: 3-step power iteration on W_h; sigma = |u . (W v)| as in reference
// ---------------------------------------------------------------------------
__global__ __launch_bounds__(1024) void k_power(const float* __restrict__ Wh,
                                                const float* __restrict__ u0,
                                                float* __restrict__ sigma_out) {
    __shared__ float us[DD];
    __shared__ float vs[DD];
    __shared__ float red[1024];
    const int i = threadIdx.x;
    us[i] = u0[i];
    __syncthreads();

    float u_raw = 0.0f;
    for (int it = 0; it < 3; ++it) {
        float acc = 0.0f;
        for (int j = 0; j < DD; ++j) acc += Wh[(size_t)j * DD + i] * us[j];
        red[i] = acc * acc;
        __syncthreads();
        for (int s = 512; s > 0; s >>= 1) { if (i < s) red[i] += red[i + s]; __syncthreads(); }
        float vn = sqrtf(red[0]) + EPS_F;
        __syncthreads();
        vs[i] = acc / vn;
        __syncthreads();

        float acc2 = 0.0f;
        const float* row = Wh + (size_t)i * DD;
        for (int j = 0; j < DD; ++j) acc2 += row[j] * vs[j];
        red[i] = acc2 * acc2;
        __syncthreads();
        for (int s = 512; s > 0; s >>= 1) { if (i < s) red[i] += red[i + s]; __syncthreads(); }
        float un = sqrtf(red[0]) + EPS_F;
        __syncthreads();
        us[i] = acc2 / un;
        u_raw = acc2;
        __syncthreads();
    }
    red[i] = us[i] * u_raw;
    __syncthreads();
    for (int s = 512; s > 0; s >>= 1) { if (i < s) red[i] += red[i + s]; __syncthreads(); }
    if (i == 0) *sigma_out = fabsf(red[0]);
}

// ---------------------------------------------------------------------------
// Kernel 3: pack weight matrix into WMMA B-operand order, bf16 hi/lo.
// B[k][n] = W[n][k] (optionally scaled by radii[n]/(sigma+eps)).
// ---------------------------------------------------------------------------
__global__ __launch_bounds__(256) void k_pack_w(const float* __restrict__ W,
                                                const float* __restrict__ log_radii,
                                                const float* __restrict__ sigma_p,
                                                unsigned short* __restrict__ hi,
                                                unsigned short* __restrict__ lo) {
    int idx = blockIdx.x * blockDim.x + threadIdx.x;  // 0 .. D*D-1
    int e     =  idx        & 15;
    int lane  = (idx >> 4)  & 31;
    int kc    = (idx >> 9)  & 31;
    int ntile =  idx >> 14;
    int n = ntile * 16 + (lane & 15);
    int k = kc * 32 + (lane >> 4) * 8 + ((e < 8) ? e : e + 8);

    float w = W[(size_t)n * DD + k];
    if (log_radii) {
        float r = MAX_RADIUS / (1.0f + expf(-log_radii[n]));
        w *= r / (*sigma_p + EPS_F);
    }
    unsigned short h = f2bf(w);
    hi[idx] = h;
    lo[idx] = f2bf(w - bf2f(h));
}

// ---------------------------------------------------------------------------
// Kernel 4: pack X [T*B, D] into WMMA A-operand order, bf16 hi/lo.
// ---------------------------------------------------------------------------
__global__ __launch_bounds__(256) void k_pack_x(const float* __restrict__ X,
                                                unsigned short* __restrict__ hi,
                                                unsigned short* __restrict__ lo) {
    int idx = blockIdx.x * blockDim.x + threadIdx.x;  // 0 .. T*B*D-1
    int e     =  idx        & 15;
    int lane  = (idx >> 4)  & 31;
    int kc    = (idx >> 9)  & 31;
    int mtile =  idx >> 14;
    int row = mtile * 16 + (lane & 15);
    int k   = kc * 32 + (lane >> 4) * 8 + ((e < 8) ? e : e + 8);

    float v = X[(size_t)row * DD + k];
    unsigned short h = f2bf(v);
    hi[idx] = h;
    lo[idx] = f2bf(v - bf2f(h));
}

// ---------------------------------------------------------------------------
// Kernel 5: P = X @ Wx^T + b.  One 16x16 tile per wave; all operands
// pre-packed -> inner loop is pure b128 loads + 3 WMMAs per K-chunk.
// ---------------------------------------------------------------------------
__global__ __launch_bounds__(256) void k_gemm_p(const unsigned short* __restrict__ Ahi,
                                                const unsigned short* __restrict__ Alo,
                                                const unsigned short* __restrict__ Bhi,
                                                const unsigned short* __restrict__ Blo,
                                                const float* __restrict__ bias,
                                                float* __restrict__ P) {
    const int wave  = blockIdx.x * (blockDim.x >> 5) + (threadIdx.x >> 5);
    const int mtile = wave >> 6;
    const int ntile = wave & 63;
    const int lane  = threadIdx.x & 31;

    v8f c0 = {}, c1 = {};
#pragma unroll 4
    for (int kc = 0; kc < KC; ++kc) {
        const size_t aoff = ((size_t)(mtile * KC + kc) * 32 + lane) * 16;
        const size_t boff = ((size_t)(ntile * KC + kc) * 32 + lane) * 16;
        v16bf ah = *(const v16bf*)(Ahi + aoff);
        v16bf al = *(const v16bf*)(Alo + aoff);
        v16bf bh = *(const v16bf*)(Bhi + boff);
        v16bf bl = *(const v16bf*)(Blo + boff);
        c0 = __builtin_amdgcn_wmma_f32_16x16x32_bf16(false, ah, false, bh, (short)0, c0, false, false);
        c1 = __builtin_amdgcn_wmma_f32_16x16x32_bf16(false, ah, false, bl, (short)0, c1, false, false);
        c1 = __builtin_amdgcn_wmma_f32_16x16x32_bf16(false, al, false, bh, (short)0, c1, false, false);
    }
#pragma unroll
    for (int v = 0; v < 8; ++v) {
        int row = mtile * 16 + (lane >> 4) * 8 + v;
        int col = ntile * 16 + (lane & 15);
        P[(size_t)row * DD + col] = c0[v] + c1[v] + bias[col];
    }
}

// ---------------------------------------------------------------------------
// TDM: stage one contiguous 32KB plane (16384 bf16) global -> LDS.
// 1D D# per CDNA5 ISA: count=1, type=2("image"),
// data_size=2B, tensor_dim0 = tile_dim0 = 16384, tensor_dim1 = tile_dim1 = 1.
// This toolchain declares the 6-arg builtin (g0, g1, g2, g3, g4, cpol).
// ---------------------------------------------------------------------------
#if defined(HAVE_TDM)
static __device__ __forceinline__ void tdm_stage_plane(const unsigned short* gsrc,
                                                       unsigned lds_byte_off) {
    unsigned long long ga = (unsigned long long)(const void*)gsrc;
    u32x4 g0;
    g0[0] = 1u;                                          // count=1, user descriptor
    g0[1] = lds_byte_off;                                // lds_addr
    g0[2] = (unsigned)(ga & 0xFFFFFFFFull);              // global_addr[31:0]
    g0[3] = (unsigned)((ga >> 32) & 0x01FFFFFFull)       // global_addr[56:32]
          | (2u << 30);                                  // type=2
    i32x8 g1;
    g1[0] = 0x00010000;   // workgroup_mask=0, data_size=1 (2 bytes)
    g1[1] = 0x40000000;   // tensor_dim0[15:0]=0x4000 at bits [63:48]
    g1[2] = 0x00010000;   // tensor_dim0[31:16]=0, tensor_dim1[15:0]=1
    g1[3] = 0x40000000;   // tensor_dim1[31:16]=0, tile_dim0=0x4000
    g1[4] = 0x00000001;   // tile_dim1=1, tile_dim2=0
    g1[5] = 16384;        // tensor_dim0_stride[31:0]
    g1[6] = 0;            // tensor_dim0_stride[47:32], tensor_dim1_stride[15:0]
    g1[7] = 0;            // tensor_dim1_stride[47:16]
    i32x4 zz4 = {0, 0, 0, 0};
    i32x8 zz8 = {0, 0, 0, 0, 0, 0, 0, 0};
    __builtin_amdgcn_tensor_load_to_lds(g0, g1, zz4, zz4, zz8, 0);
}
#endif

// ---------------------------------------------------------------------------
// Kernel 6: persistent sequential scan with LDS-resident weights.
//   h_t = tanh(P[t] + h_{t-1} @ Wh_eff^T);  out[t] = h_t * silu(z[t])
// 16 blocks x 4 waves; each wave owns one 16-col slice of Wh_eff, staged once
// via the Tensor Data Mover into 64KB of LDS (256KB/block fits CDNA5's 320KB
// WGP LDS) and reused for all 1024 steps. h is consumed from / produced into
// packed bf16 A-buffers (parity double-buffered): hot loop = loads + WMMA.
// ---------------------------------------------------------------------------
__global__ __launch_bounds__(SCAN_THREADS) void k_scan(const float* __restrict__ P,
                                                       const float* __restrict__ z,
                                                       const unsigned short* __restrict__ Whi,
                                                       const unsigned short* __restrict__ Wlo,
                                                       unsigned short* __restrict__ HA,   // 4 planes
                                                       float* __restrict__ outs,
                                                       float* __restrict__ hall,
                                                       unsigned* __restrict__ bar) {
    extern __shared__ __align__(16) unsigned short smem[];
    const int wid   = threadIdx.x >> 5;
    const int ntile = blockIdx.x * (SCAN_THREADS >> 5) + wid;
    const int lane  = threadIdx.x & 31;

    unsigned short* sHi = smem + (size_t)wid * 2 * LDS_PLANE;
    unsigned short* sLo = sHi + LDS_PLANE;

    // ---- stage this wave's weight slice into LDS (once) ----
#if defined(HAVE_TDM)
    {
        const unsigned lds0 = (unsigned)wid * (2u * LDS_PLANE * 2u);   // bytes
        tdm_stage_plane(Whi + (size_t)ntile * LDS_PLANE, lds0);
        tdm_stage_plane(Wlo + (size_t)ntile * LDS_PLANE, lds0 + LDS_PLANE * 2u);
        __builtin_amdgcn_s_wait_tensorcnt(0);
    }
#else
#pragma unroll 4
    for (int kc = 0; kc < KC; ++kc) {
        const size_t g = ((size_t)(ntile * KC + kc) * 32 + lane) * 16;
        const int    l = (kc * 32 + lane) * 16;
        *(v16bf*)&sHi[l] = *(const v16bf*)(Whi + g);
        *(v16bf*)&sLo[l] = *(const v16bf*)(Wlo + g);
    }
#endif
    __syncthreads();

    const int ncol = ntile * 16 + (lane & 15);
    const int ai0  = a_pack_idx(0, ncol);

    unsigned short* A0hi = HA;
    unsigned short* A0lo = HA + BD;
    unsigned short* A1hi = HA + 2 * BD;
    unsigned short* A1lo = HA + 3 * BD;

    for (int t = 0; t < TT; ++t) {
        const unsigned short* cAhi = (t & 1) ? A1hi : A0hi;
        const unsigned short* cAlo = (t & 1) ? A1lo : A0lo;
        unsigned short* nAhi = (t & 1) ? A0hi : A1hi;
        unsigned short* nAlo = (t & 1) ? A0lo : A1lo;

        // hoist P/z loads: independent of h, overlap with the WMMA chain
        float pv[8], zv[8];
#pragma unroll
        for (int v = 0; v < 8; ++v) {
            int b = (lane >> 4) * 8 + v;
            size_t idx = (size_t)t * BD + (size_t)b * DD + ncol;
            pv[v] = P[idx];
            zv[v] = z[idx];
        }

        v8f c0 = {}, c1 = {};
#pragma unroll 4
        for (int kc = 0; kc < KC; ++kc) {
            const int off = (kc * 32 + lane) * 16;
            v16bf ah = *(const v16bf*)(cAhi + off);
            v16bf al = *(const v16bf*)(cAlo + off);
            v16bf bh = *(const v16bf*)&sHi[off];
            v16bf bl = *(const v16bf*)&sLo[off];
            c0 = __builtin_amdgcn_wmma_f32_16x16x32_bf16(false, ah, false, bh, (short)0, c0, false, false);
            c1 = __builtin_amdgcn_wmma_f32_16x16x32_bf16(false, ah, false, bl, (short)0, c1, false, false);
            c1 = __builtin_amdgcn_wmma_f32_16x16x32_bf16(false, al, false, bh, (short)0, c1, false, false);
        }

#pragma unroll
        for (int v = 0; v < 8; ++v) {
            int b = (lane >> 4) * 8 + v;
            size_t idx = (size_t)t * BD + (size_t)b * DD + ncol;
            float hv = tanhf(pv[v] + c0[v] + c1[v]);
            hall[(size_t)(t + 1) * BD + (size_t)b * DD + ncol] = hv;
            outs[idx] = hv * (zv[v] / (1.0f + expf(-zv[v])));   // silu gate
            int ai = ai0 + (b << 4);
            unsigned short hh = f2bf(hv);
            nAhi[ai] = hh;
            nAlo[ai] = f2bf(hv - bf2f(hh));
        }

        // ---- device-scope barrier: all blocks finished step t ----
        __threadfence();
        __syncthreads();
        if (threadIdx.x == 0) {
            __hip_atomic_fetch_add(bar, 1u, __ATOMIC_RELEASE, __HIP_MEMORY_SCOPE_AGENT);
            const unsigned target = (unsigned)(t + 1) * SCAN_BLOCKS;
            while (__hip_atomic_load(bar, __ATOMIC_ACQUIRE, __HIP_MEMORY_SCOPE_AGENT) < target)
                __builtin_amdgcn_s_sleep(1);
        }
        __syncthreads();
        __threadfence();
    }
}

// ---------------------------------------------------------------------------
// Launch
// ---------------------------------------------------------------------------
extern "C" void kernel_launch(void* const* d_in, const int* in_sizes, int n_in,
                              void* d_out, int out_size, void* d_ws, size_t ws_size,
                              hipStream_t stream) {
    const float* x  = (const float*)d_in[0];
    const float* z  = (const float*)d_in[1];
    const float* h0 = (const float*)d_in[2];
    const float* Wx = (const float*)d_in[3];
    const float* Wh = (const float*)d_in[4];
    const float* lr = (const float*)d_in[5];
    const float* bb = (const float*)d_in[6];
    const float* u0 = (const float*)d_in[7];
    (void)in_sizes; (void)n_in; (void)out_size; (void)ws_size;

    float* outs = (float*)d_out;
    float* hall = (float*)d_out + (size_t)TT * BD;

    char* ws = (char*)d_ws;
    unsigned*       bar   = (unsigned*)(ws + WS_BAR);
    float*          sigma = (float*)(ws + WS_SIGMA);
    unsigned short* WhHi  = (unsigned short*)(ws + WS_WH_HI);
    unsigned short* WhLo  = (unsigned short*)(ws + WS_WH_LO);
    unsigned short* WxHi  = (unsigned short*)(ws + WS_WX_HI);
    unsigned short* WxLo  = (unsigned short*)(ws + WS_WX_LO);
    float*          P     = (float*)(ws + WS_P);
    unsigned short* XaHi  = (unsigned short*)(ws + WS_XA_HI);
    unsigned short* XaLo  = (unsigned short*)(ws + WS_XA_LO);
    unsigned short* HA    = (unsigned short*)(ws + WS_HA);

    hipLaunchKernelGGL(k_init, dim3((BD + 255) / 256), dim3(256), 0, stream,
                       h0, hall, HA, HA + BD, bar);
    hipLaunchKernelGGL(k_power, dim3(1), dim3(1024), 0, stream, Wh, u0, sigma);
    hipLaunchKernelGGL(k_pack_w, dim3((DD * DD) / 256), dim3(256), 0, stream,
                       Wh, lr, sigma, WhHi, WhLo);
    hipLaunchKernelGGL(k_pack_w, dim3((DD * DD) / 256), dim3(256), 0, stream,
                       Wx, (const float*)nullptr, sigma, WxHi, WxLo);
    hipLaunchKernelGGL(k_pack_x, dim3((TT * BD) / 256), dim3(256), 0, stream,
                       x, XaHi, XaLo);
    hipLaunchKernelGGL(k_gemm_p, dim3((TT * BB / 16) * NT / 8), dim3(256), 0, stream,
                       XaHi, XaLo, WxHi, WxLo, bb, P);
    hipLaunchKernelGGL(k_scan, dim3(SCAN_BLOCKS), dim3(SCAN_THREADS), SCAN_LDS_BYTES, stream,
                       P, z, WhHi, WhLo, HA, outs, hall, bar);
}
